// WeightAttention_53017076302310
// MI455X (gfx1250) — compile-verified, matrix operand order checked by reading, not run on hardware
//
#include <hip/hip_runtime.h>
#include <hip/hip_bf16.h>
#include <math.h>

typedef __attribute__((ext_vector_type(16))) __bf16 v16bf;
typedef __attribute__((ext_vector_type(8)))  __bf16 v8bf;
typedef __attribute__((ext_vector_type(8)))  float  v8f;
typedef __attribute__((ext_vector_type(4)))  float  v4f;

#define DIM   1024
#define NH    16
#define HD    64
#define BATCH 4
#define QN    1024
#define KN    1024
#define PROWS 8192   // 4096 projected-query rows + 4096 projected-key rows

#define WMMA_BF16(A, B, C) \
  __builtin_amdgcn_wmma_f32_16x16x32_bf16(false, (A), false, (B), (short)0, (C), false, false)

// A light fence between the two ping-pong halves: stops the scheduler from
// hoisting the set-0 prefetch loads above the set-0 WMMA group (which forced
// temp-register copies), without imposing a rigid full-body schedule (which
// made the allocator spill accumulators).
#if __has_builtin(__builtin_amdgcn_sched_barrier)
#define SCHED_FENCE() __builtin_amdgcn_sched_barrier(0)
#else
#define SCHED_FENCE() do { } while (0)
#endif

// ---------------------------------------------------------------------------
// Operand loader for v_wmma_f32_16x16x32_bf16 (wave32), ISA 7.12.2 layout:
// lane = {half,row}; elements j=0..7 hold K = 8*half + j, j=8..15 hold
// K = 16 + 8*half + (j-8). Same per-lane K map for A (M=lane&15) and B
// (N=lane&15) since both operands stream along K (row-major K-contiguous).
// ---------------------------------------------------------------------------
__device__ __forceinline__ v16bf load_frag(const __bf16* __restrict__ base,
                                           int g0, int g1) {
  v8bf lo = *(const v8bf*)(base + g0);
  v8bf hi = *(const v8bf*)(base + g1);
  v16bf r;
#pragma unroll
  for (int j = 0; j < 8; ++j) { r[j] = lo[j]; r[j + 8] = hi[j]; }
  return r;
}

// ---------------------------------------------------------------------------
// Kernel 0: fp32 -> bf16 bulk convert (read-once data: non-temporal loads).
// ---------------------------------------------------------------------------
__global__ void cvt_kernel(const float* __restrict__ src,
                           __bf16* __restrict__ dst, int n8) {
  int i = blockIdx.x * blockDim.x + threadIdx.x;
  if (i >= n8) return;
  const v4f* s = (const v4f*)src + (size_t)i * 2;
  v4f x0 = __builtin_nontemporal_load(s);
  v4f x1 = __builtin_nontemporal_load(s + 1);
  v8bf o;
#pragma unroll
  for (int j = 0; j < 4; ++j) { o[j] = (__bf16)x0[j]; o[j + 4] = (__bf16)x1[j]; }
  *((v8bf*)dst + i) = o;
}

// ---------------------------------------------------------------------------
// One K=1024 GEMM pass for a 16(M) x 64(N) tile: A row tile `arow` against
// 4 consecutive B row tiles starting at `brow` (spaced 16*DIM).
// Ping-pong double buffering; a sched fence between halves keeps each
// buffer set's loads behind the other set's WMMAs (no rotation copies).
// ---------------------------------------------------------------------------
__device__ __forceinline__ void gemm_pass(const __bf16* __restrict__ arow,
                                          const __bf16* __restrict__ brow,
                                          int g0, int g1,
                                          v8f& d0, v8f& d1, v8f& d2, v8f& d3) {
  const __bf16* br0 = brow;
  const __bf16* br1 = brow + (size_t)16 * DIM;
  const __bf16* br2 = brow + (size_t)32 * DIM;
  const __bf16* br3 = brow + (size_t)48 * DIM;

  // buffer set 0 primed at k = 0
  v16bf a0  = load_frag(arow, g0, g1);
  v16bf p00 = load_frag(br0, g0, g1);
  v16bf p01 = load_frag(br1, g0, g1);
  v16bf p02 = load_frag(br2, g0, g1);
  v16bf p03 = load_frag(br3, g0, g1);

#pragma unroll 1
  for (int k = 0; k < DIM - 64; k += 64) {     // k = 0, 64, ..., 896
    // prefetch set 1 @ k+32, consume set 0 @ k
    v16bf a1  = load_frag(arow + k + 32, g0, g1);
    v16bf p10 = load_frag(br0 + k + 32, g0, g1);
    v16bf p11 = load_frag(br1 + k + 32, g0, g1);
    v16bf p12 = load_frag(br2 + k + 32, g0, g1);
    v16bf p13 = load_frag(br3 + k + 32, g0, g1);
    d0 = WMMA_BF16(a0, p00, d0);
    d1 = WMMA_BF16(a0, p01, d1);
    d2 = WMMA_BF16(a0, p02, d2);
    d3 = WMMA_BF16(a0, p03, d3);
    SCHED_FENCE();
    // prefetch set 0 @ k+64, consume set 1 @ k+32
    a0  = load_frag(arow + k + 64, g0, g1);
    p00 = load_frag(br0 + k + 64, g0, g1);
    p01 = load_frag(br1 + k + 64, g0, g1);
    p02 = load_frag(br2 + k + 64, g0, g1);
    p03 = load_frag(br3 + k + 64, g0, g1);
    d0 = WMMA_BF16(a1, p10, d0);
    d1 = WMMA_BF16(a1, p11, d1);
    d2 = WMMA_BF16(a1, p12, d2);
    d3 = WMMA_BF16(a1, p13, d3);
  }
  // tail: set 0 holds k = DIM-64; last stage k = DIM-32 into set 1
  {
    v16bf a1  = load_frag(arow + DIM - 32, g0, g1);
    v16bf p10 = load_frag(br0 + DIM - 32, g0, g1);
    v16bf p11 = load_frag(br1 + DIM - 32, g0, g1);
    v16bf p12 = load_frag(br2 + DIM - 32, g0, g1);
    v16bf p13 = load_frag(br3 + DIM - 32, g0, g1);
    d0 = WMMA_BF16(a0, p00, d0);
    d1 = WMMA_BF16(a0, p01, d1);
    d2 = WMMA_BF16(a0, p02, d2);
    d3 = WMMA_BF16(a0, p03, d3);
    d0 = WMMA_BF16(a1, p10, d0);
    d1 = WMMA_BF16(a1, p11, d1);
    d2 = WMMA_BF16(a1, p12, d2);
    d3 = WMMA_BF16(a1, p13, d3);
  }
}

// ---------------------------------------------------------------------------
// Kernel 1: P[8192x1024](bf16) = [query;key] @ W^T + b, all-bf16 operands.
// ---------------------------------------------------------------------------
__device__ __forceinline__ void store_proj_tile(v8f acc, float bv,
                                                __bf16* __restrict__ P,
                                                int prow0, int col, int half) {
#pragma unroll
  for (int r = 0; r < 8; ++r) {
    int m = r + 8 * half;                    // C layout: VGPR r -> M = r + 8*half
    P[(size_t)(prow0 + m) * DIM + col] = (__bf16)(acc[r] + bv);
  }
}

__global__ void proj_kernel(const __bf16* __restrict__ Xbf,
                            const __bf16* __restrict__ Wbf,
                            const float* __restrict__ bias,
                            __bf16* __restrict__ P) {
  const int wave = (blockIdx.x * blockDim.x + threadIdx.x) >> 5; // 0..8191
  const int tm   = wave >> 4;        // 0..511  (16 rows of P)
  const int ng   = wave & 15;        // 0..15   (64 output channels)
  const int lane = threadIdx.x & 31;
  const int half = lane >> 4;
  const int rw   = lane & 15;
  const int g0   = 8 * half;
  const int g1   = 16 + 8 * half;

  const __bf16* xrow = Xbf + (size_t)(tm * 16 + rw) * DIM;
  const __bf16* wrow = Wbf + (size_t)(ng * 64 + rw) * DIM;

  v8f acc0 = {}, acc1 = {}, acc2 = {}, acc3 = {};
  gemm_pass(xrow, wrow, g0, g1, acc0, acc1, acc2, acc3);

  const int prow0 = tm * 16;
  const int col0  = ng * 64 + rw;
  store_proj_tile(acc0, bias[col0],      P, prow0, col0,      half);
  store_proj_tile(acc1, bias[col0 + 16], P, prow0, col0 + 16, half);
  store_proj_tile(acc2, bias[col0 + 32], P, prow0, col0 + 32, half);
  store_proj_tile(acc3, bias[col0 + 48], P, prow0, col0 + 48, half);
}

// ---------------------------------------------------------------------------
// Kernel 2: per-(row, head) normalize: Pn[row, h*64+d] = P / ||P_head||.
// Then  sum_h q_h.k_h / (|q_h||k_h|)  ==  full-K dot of Pn rows,
// and   sum_h q_h.k_h                ==  full-K dot of P rows.
// ---------------------------------------------------------------------------
__global__ void normize_kernel(const __bf16* __restrict__ P,
                               __bf16* __restrict__ Pn) {
  int id = blockIdx.x * blockDim.x + threadIdx.x;
  if (id >= PROWS * NH) return;
  int row = id >> 4, h = id & 15;
  const __bf16* p = P  + (size_t)row * DIM + h * HD;
  __bf16*       q = Pn + (size_t)row * DIM + h * HD;
  v8bf v[8];
  float s = 0.f;
#pragma unroll
  for (int i = 0; i < 8; ++i) {
    v[i] = *(const v8bf*)(p + i * 8);
#pragma unroll
    for (int j = 0; j < 8; ++j) { float f = (float)v[i][j]; s = fmaf(f, f, s); }
  }
  float inv = (s > 1e-30f) ? rsqrtf(s) : 0.f;   // guards cos = S/max(.,1e-8)
#pragma unroll
  for (int i = 0; i < 8; ++i) {
    v8bf o;
#pragma unroll
    for (int j = 0; j < 8; ++j) o[j] = (__bf16)((float)v[i][j] * inv);
    *(v8bf*)(q + i * 8) = o;
  }
}

// ---------------------------------------------------------------------------
// Kernel 3: two K=1024 GEMM passes (raw P -> weight_mm, normalized Pn ->
// weight_cos) + mask + head-mean + relu. One wave per 16(M) x 64(N) tile.
// ---------------------------------------------------------------------------
__device__ __forceinline__ void store_out_tile(v8f sm, v8f sc,
                                               const int* __restrict__ mask,
                                               float* __restrict__ out_cos,
                                               float* __restrict__ out_mm,
                                               int b, int qt, int kcol, int half) {
  const bool msk = mask[b * KN + kcol] != 0;
#pragma unroll
  for (int r = 0; r < 8; ++r) {
    const int    qrow_g = qt * 16 + r + 8 * half;
    const size_t o = ((size_t)b * QN + qrow_g) * KN + kcol;
    float vc = msk ? 0.0f : fmaxf(sc[r] * (1.0f / NH), 0.0f);
    float vm = msk ? -INFINITY : sm[r] * (1.0f / NH);
    __builtin_nontemporal_store(vc, out_cos + o);
    __builtin_nontemporal_store(vm, out_mm + o);
  }
}

__global__ void score_kernel(const __bf16* __restrict__ P,
                             const __bf16* __restrict__ Pn,
                             const int* __restrict__ mask,
                             float* __restrict__ out_cos,
                             float* __restrict__ out_mm) {
  const int wave = (blockIdx.x * blockDim.x + threadIdx.x) >> 5; // 0..4095
  const int b    = wave >> 10;           // /1024
  const int qt   = (wave >> 4) & 63;     // 64 q-tiles
  const int kg   = wave & 15;            // 16 k-groups of 64
  const int lane = threadIdx.x & 31;
  const int half = lane >> 4;
  const int rw   = lane & 15;
  const int g0   = 8 * half;
  const int g1   = 16 + 8 * half;

  const size_t qoff = (size_t)(b * QN + qt * 16 + rw) * DIM;
  const size_t koff = (size_t)(4096 + b * KN + kg * 64 + rw) * DIM;

  v8f am0 = {}, am1 = {}, am2 = {}, am3 = {};
  gemm_pass(P + qoff, P + koff, g0, g1, am0, am1, am2, am3);

  v8f ac0 = {}, ac1 = {}, ac2 = {}, ac3 = {};
  gemm_pass(Pn + qoff, Pn + koff, g0, g1, ac0, ac1, ac2, ac3);

  const int kcol0 = kg * 64 + rw;
  store_out_tile(am0, ac0, mask, out_cos, out_mm, b, qt, kcol0,      half);
  store_out_tile(am1, ac1, mask, out_cos, out_mm, b, qt, kcol0 + 16, half);
  store_out_tile(am2, ac2, mask, out_cos, out_mm, b, qt, kcol0 + 32, half);
  store_out_tile(am3, ac3, mask, out_cos, out_mm, b, qt, kcol0 + 48, half);
}

// ---------------------------------------------------------------------------
extern "C" void kernel_launch(void* const* d_in, const int* in_sizes, int n_in,
                              void* d_out, int out_size, void* d_ws, size_t ws_size,
                              hipStream_t stream) {
  const float* query = (const float*)d_in[0];   // (4,1024,1024)
  const float* key   = (const float*)d_in[1];   // (4,1024,1024)
  const int*   mask  = (const int*)d_in[2];     // (4,1024) bool
  const float* W     = (const float*)d_in[3];   // (1024,1024)
  const float* bias  = (const float*)d_in[4];   // (1024,)

  float* out_cos = (float*)d_out;
  float* out_mm  = out_cos + (size_t)BATCH * QN * KN;

  char* ws = (char*)d_ws;
  __bf16* Xbf = (__bf16*)ws;                                  // 16 MB
  __bf16* Wbf = (__bf16*)(ws + (size_t)16 * 1024 * 1024);     //  2 MB
  __bf16* P   = (__bf16*)(ws + (size_t)18 * 1024 * 1024);     // 16 MB
  __bf16* Pn  = (__bf16*)(ws + (size_t)34 * 1024 * 1024);     // 16 MB

  // fp32 -> bf16 staging (each source converted exactly once)
  cvt_kernel<<<2048, 256, 0, stream>>>(query, Xbf, (BATCH * QN * DIM) / 8);
  cvt_kernel<<<2048, 256, 0, stream>>>(key, Xbf + (size_t)BATCH * QN * DIM,
                                       (BATCH * KN * DIM) / 8);
  cvt_kernel<<<512, 256, 0, stream>>>(W, Wbf, (DIM * DIM) / 8);

  // projection: 512 M-tiles x 16 N-groups, 1 wave each, 4 waves/block
  proj_kernel<<<2048, 128, 0, stream>>>(Xbf, Wbf, bias, P);

  // per-head normalized copy
  normize_kernel<<<(PROWS * NH + 255) / 256, 256, 0, stream>>>(P, Pn);

  // fused scores: 4 * 64 * 16 = 4096 waves, 4 waves/block
  score_kernel<<<1024, 128, 0, stream>>>(P, Pn, mask, out_cos, out_mm);
}
